// FastRCNN_73048803770557
// MI455X (gfx1250) — compile-verified, matrix operand order checked by reading, not run on hardware
//
#include <hip/hip_runtime.h>
#include <math.h>

// ---------------------------------------------------------------------------
// Fast-RCNN head for MI455X (gfx1250, wave32, WMMA).
// Dominant cost: X[1024,12544] @ W1[12544,1024]  (~26 GFLOP, 51MB weights).
// Weight-BW bound => bf16 data path + v_wmma_f32_16x16x32_bf16, f32 accum,
// async global->LDS staging (ASYNCcnt) with double buffering.
// ---------------------------------------------------------------------------

typedef __attribute__((ext_vector_type(16))) __bf16 v16bf;
typedef __attribute__((ext_vector_type(8)))  float  v8f;

#define IMGSZ      640.0f
#define NB         2
#define NP         512
#define NROI       (NB * NP)          // 1024
#define NCH        256
#define FH         40
#define FW         40
#define PP         7
#define NBIN       (PP * PP)          // 49
#define DIN        (NCH * NBIN)       // 12544
#define DH         1024
#define NCLS       81
#define NBOX       (4 * NCLS)         // 324
#define NBOX_PAD   384
#define NCLS_PAD   128
#define SCALE_F    0.0625f
#define SCORE_TH   0.05f
#define NMS_TH     0.5f
#define MIN_SIZE   0.01f
#define DET_PER    100
#define KTOP       1000
#define NCAND      (NP * (NCLS - 1))  // 40960 per image
#define BBOX_CLAMP 4.135166556742356f // log(1000/16)

#if __has_builtin(__builtin_amdgcn_global_load_async_to_lds_b128) && \
    __has_builtin(__builtin_amdgcn_s_wait_asynccnt)
#define USE_ASYNC_LDS 1
#else
#define USE_ASYNC_LDS 0
#endif

// Builtin signature (from compiler diagnostic): v4i addrspace(1)* src,
// v4i addrspace(3)* dst, imm offset, imm cpol. Cast through integers:
// AS(1) = flat 64-bit VA; AS(3) = low-32-bit LDS offset of a generic LDS ptr.
typedef int v4i_async __attribute__((ext_vector_type(4)));
typedef __attribute__((address_space(1))) v4i_async* as1_v4i;
typedef __attribute__((address_space(3))) v4i_async* as3_v4i;

// ---------------------------------------------------------------------------
// f32 -> bf16 weight conversion with N-padding (zero pad keeps GEMM guard-free)
// ---------------------------------------------------------------------------
__global__ void convert_pad_bf16(const float* __restrict__ src,
                                 __bf16* __restrict__ dst,
                                 int K, int Ns, int Npad) {
    long i = (long)blockIdx.x * blockDim.x + threadIdx.x;
    long total = (long)K * Npad;
    if (i >= total) return;
    int  n = (int)(i % Npad);
    long k = i / Npad;
    float v = (n < Ns) ? src[k * Ns + n] : 0.0f;
    dst[i] = (__bf16)v;
}

__global__ void pad_bias_f32(const float* __restrict__ src,
                             float* __restrict__ dst, int Ns, int Npad) {
    int i = blockIdx.x * blockDim.x + threadIdx.x;
    if (i < Npad) dst[i] = (i < Ns) ? src[i] : 0.0f;
}

// ---------------------------------------------------------------------------
// ROI max pool: one block per (roi, bin); 256 threads = channels.
// ---------------------------------------------------------------------------
__global__ __launch_bounds__(256) void roi_pool(const float* __restrict__ feats,
                                                const float* __restrict__ props,
                                                __bf16* __restrict__ X) {
    int blk = blockIdx.x;
    int roi = blk / NBIN;
    int bin = blk % NBIN;
    int c   = threadIdx.x;
    int b   = roi >> 9;               // 512 rois per image

    const float* bx = props + roi * 4;
    float x1 = bx[0], y1 = bx[1], x2 = bx[2], y2 = bx[3];
    int rs_w = (int)roundf(x1 * SCALE_F);
    int rs_h = (int)roundf(y1 * SCALE_F);
    int re_w = (int)roundf(x2 * SCALE_F);
    int re_h = (int)roundf(y2 * SCALE_F);
    int iw = re_w - rs_w + 1; if (iw < 1) iw = 1;
    int ih = re_h - rs_h + 1; if (ih < 1) ih = 1;
    float bin_w = (float)iw / (float)PP;
    float bin_h = (float)ih / (float)PP;
    int ph = bin / PP, pw = bin % PP;

    int hs = rs_h + (int)floorf(ph * bin_h);
    int he = rs_h + (int)ceilf((ph + 1) * bin_h);
    int ws = rs_w + (int)floorf(pw * bin_w);
    int we = rs_w + (int)ceilf((pw + 1) * bin_w);
    hs = min(max(hs, 0), FH); he = min(max(he, 0), FH);
    ws = min(max(ws, 0), FW); we = min(max(we, 0), FW);

    const float* fc = feats + (((long)b * NCH + c) * FH) * FW;
    float m = -3.0e38f;
    bool any = false;
#pragma unroll
    for (int kh = 0; kh < 8; ++kh) {
        int hi  = hs + kh;
        bool vh = hi < he;
        int hidx = min(max(hi, 0), FH - 1);
#pragma unroll
        for (int kw = 0; kw < 8; ++kw) {
            int wi  = ws + kw;
            bool vw = wi < we;
            if (vh && vw) {
                float v = fc[hidx * FW + min(max(wi, 0), FW - 1)];
                m = fmaxf(m, v);
                any = true;
            }
        }
    }
    X[(long)roi * DIN + c * NBIN + bin] = (__bf16)(any ? m : 0.0f);
}

// ---------------------------------------------------------------------------
// bf16 WMMA GEMM: C = act(A[MxK] @ B[KxN] + bias)
// Block = 256 threads = 8 wave32 waves; 64x64 tile; K-step 32; double-buffered
// LDS. A tile staged with GLOBAL_LOAD_ASYNC_TO_LDS_B128 (ASYNCcnt path) when
// available; B tile staged through registers with a transposed store so both
// fragment gathers become two contiguous ds_load_b128 runs per lane
// (16-bit A layout: elems 0..7 -> K=kb+e, 8..15 -> K=16+kb+e, kb=(lane>>4)*8).
// ---------------------------------------------------------------------------
__global__ __launch_bounds__(256) void gemm_wmma_bf16(const __bf16* __restrict__ A,
                                                      const __bf16* __restrict__ Bm,
                                                      const float*  __restrict__ bias,
                                                      float*   Cf,
                                                      __bf16*  Cb,
                                                      int M, int N, int K, int relu) {
    __shared__ __bf16 As [2][64][32];
    __shared__ __bf16 Bts[2][64][32];

    const int tid  = threadIdx.x;
    const int lane = tid & 31;
    const int wave = tid >> 5;
    const int mt   = wave & 3;
    const int ntb  = (wave >> 2) << 1;
    const int n0   = blockIdx.x * 64;
    const int m0   = blockIdx.y * 64;

    // per-thread staging coordinates
    const int rowA = tid >> 2;            // 0..63
    const int colA = (tid & 3) * 8;       // 0,8,16,24  (16B chunk)
    const int kkB  = tid >> 3;            // 0..31
    const int nnB  = (tid & 7) * 8;       // 0..56      (16B chunk)

    v8f acc0 = {};
    v8f acc1 = {};

    const int rA = mt * 16 + (lane & 15);
    const int kb = (lane >> 4) * 8;
    const int c0 = ntb * 16 + (lane & 15);
    const int c1 = c0 + 16;

    auto stage_A = [&](int buf, int k0) {
        const __bf16* src = A + (long)(m0 + rowA) * K + k0 + colA;
        __bf16* dst = &As[buf][rowA][colA];
#if USE_ASYNC_LDS
        __builtin_amdgcn_global_load_async_to_lds_b128(
            (as1_v4i)(unsigned long long)src,
            (as3_v4i)(unsigned int)(unsigned long long)dst,
            0, 0);
#else
#pragma unroll
        for (int j = 0; j < 8; ++j) dst[j] = src[j];
#endif
    };
    auto load_B = [&](__bf16* breg, int k0) {
        const __bf16* src = Bm + (long)(k0 + kkB) * N + n0 + nnB;
#pragma unroll
        for (int j = 0; j < 8; ++j) breg[j] = src[j];
    };
    auto store_B = [&](int buf, const __bf16* breg) {
#pragma unroll
        for (int j = 0; j < 8; ++j) Bts[buf][nnB + j][kkB] = breg[j];
    };
    auto wait_stage = [&]() {
#if USE_ASYNC_LDS
        __builtin_amdgcn_s_wait_asynccnt(0);
#endif
        __syncthreads();
    };

    // prologue: stage k0 = 0 into buffer 0
    __bf16 breg[8];
    stage_A(0, 0);
    load_B(breg, 0);
    store_B(0, breg);
    wait_stage();

    int buf = 0;
    for (int k0 = 0; k0 < K; k0 += 32) {
        const int  nxt  = buf ^ 1;
        const bool more = (k0 + 32) < K;
        if (more) {
            stage_A(nxt, k0 + 32);          // async: overlaps with WMMA below
            load_B(breg, k0 + 32);
            __builtin_prefetch(&Bm[(long)(k0 + 64) * N + n0 + lane], 0, 1);
        }

        v16bf a, b0, b1;
#pragma unroll
        for (int j = 0; j < 8; ++j) {
            a [j]     = As [buf][rA][kb + j];   a [j + 8] = As [buf][rA][16 + kb + j];
            b0[j]     = Bts[buf][c0][kb + j];   b0[j + 8] = Bts[buf][c0][16 + kb + j];
            b1[j]     = Bts[buf][c1][kb + j];   b1[j + 8] = Bts[buf][c1][16 + kb + j];
        }
        acc0 = __builtin_amdgcn_wmma_f32_16x16x32_bf16(false, a, false, b0,
                                                       (short)0, acc0, false, false);
        acc1 = __builtin_amdgcn_wmma_f32_16x16x32_bf16(false, a, false, b1,
                                                       (short)0, acc1, false, false);

        if (more) store_B(nxt, breg);
        wait_stage();
        buf = nxt;
    }

    // epilogue: C/D layout -> n = lane&15, m = r + 8*(lane>>4)
    const int nlocal = lane & 15;
    const int mhi    = (lane >> 4) * 8;
#pragma unroll
    for (int r = 0; r < 8; ++r) {
        int m    = m0 + mt * 16 + r + mhi;
        int col0 = n0 + ntb * 16 + nlocal;
        int col1 = col0 + 16;
        float v0 = acc0[r] + bias[col0];
        float v1 = acc1[r] + bias[col1];
        if (relu) { v0 = fmaxf(v0, 0.0f); v1 = fmaxf(v1, 0.0f); }
        if (Cb) {
            Cb[(long)m * N + col0] = (__bf16)v0;
            Cb[(long)m * N + col1] = (__bf16)v1;
        }
        if (Cf) {
            Cf[(long)m * N + col0] = v0;
            Cf[(long)m * N + col1] = v1;
        }
    }
}

// ---------------------------------------------------------------------------
// Softmax over 81 logits + box decode; one block per ROI.
// ---------------------------------------------------------------------------
__global__ __launch_bounds__(128) void decode_softmax(const float* __restrict__ boxout,
                                                      const float* __restrict__ clsout,
                                                      const float* __restrict__ props,
                                                      float* __restrict__ candb,
                                                      float* __restrict__ cands) {
    __shared__ float lg[NCLS];
    __shared__ float pr[NCLS];
    int ri = blockIdx.x;
    int b  = ri >> 9;
    int p  = ri & 511;
    int tid = threadIdx.x;

    if (tid < NCLS) lg[tid] = clsout[(long)ri * NCLS_PAD + tid];
    __syncthreads();
    if (tid == 0) {
        float mx = -3.0e38f;
        for (int c = 0; c < NCLS; ++c) mx = fmaxf(mx, lg[c]);
        float s = 0.0f;
        for (int c = 0; c < NCLS; ++c) { pr[c] = expf(lg[c] - mx); s += pr[c]; }
        float inv = 1.0f / s;
        for (int c = 0; c < NCLS; ++c) pr[c] *= inv;
    }
    __syncthreads();

    if (tid < NCLS - 1) {
        int c = tid + 1;
        const float* bx = props + ri * 4;
        float w  = bx[2] - bx[0], h = bx[3] - bx[1];
        float cx = bx[0] + 0.5f * w, cy = bx[1] + 0.5f * h;
        const float* t = boxout + (long)ri * NBOX_PAD + c * 4;
        float dx = t[0], dy = t[1];
        float dw = fminf(t[2], BBOX_CLAMP), dh = fminf(t[3], BBOX_CLAMP);
        float pcx = dx * w + cx, pcy = dy * h + cy;
        float pw  = expf(dw) * w, ph = expf(dh) * h;
        float bx1 = fminf(fmaxf(pcx - 0.5f * pw, 0.0f), IMGSZ);
        float by1 = fminf(fmaxf(pcy - 0.5f * ph, 0.0f), IMGSZ);
        float bx2 = fminf(fmaxf(pcx + 0.5f * pw, 0.0f), IMGSZ);
        float by2 = fminf(fmaxf(pcy + 0.5f * ph, 0.0f), IMGSZ);
        long ci = (long)b * NCAND + (long)p * (NCLS - 1) + (c - 1);
        candb[ci * 4 + 0] = bx1; candb[ci * 4 + 1] = by1;
        candb[ci * 4 + 2] = bx2; candb[ci * 4 + 3] = by2;
        float sc = pr[c];
        bool valid = (sc > SCORE_TH) && ((bx2 - bx1) >= MIN_SIZE) && ((by2 - by1) >= MIN_SIZE);
        cands[ci] = valid ? sc : -1.0f;
    }
}

// ---------------------------------------------------------------------------
// Per-image top-1000 (destructive iterative argmax) + greedy NMS + compaction.
// d_out layout: boxes[2*100*4] | scores[2*100] | labels[2*100]
// ---------------------------------------------------------------------------
__global__ __launch_bounds__(256) void nms_image(const float* __restrict__ candb,
                                                 float* __restrict__ cands,
                                                 float* __restrict__ out) {
    __shared__ float tb[KTOP * 4];
    __shared__ float ts[KTOP];
    __shared__ int   tl[KTOP];
    __shared__ float nbx[KTOP * 4];
    __shared__ float area[KTOP];
    __shared__ unsigned char keep[KTOP];
    __shared__ float rv[256];
    __shared__ int   rix[256];
    __shared__ int   sup;

    const int b = blockIdx.x;
    const int tid = threadIdx.x;
    const float* cb = candb + (long)b * NCAND * 4;
    float* cs = cands + (long)b * NCAND;

    // ---- top-K by iterative argmax (mark chosen with -2) ----
    for (int t = 0; t < KTOP; ++t) {
        float best = -3.0e38f; int bi = NCAND;
        for (int i = tid; i < NCAND; i += 256) {
            float v = cs[i];
            if (v > best) { best = v; bi = i; }
        }
        rv[tid] = best; rix[tid] = bi;
        __syncthreads();
        for (int s = 128; s > 0; s >>= 1) {
            if (tid < s) {
                if (rv[tid + s] > rv[tid] ||
                    (rv[tid + s] == rv[tid] && rix[tid + s] < rix[tid])) {
                    rv[tid] = rv[tid + s]; rix[tid] = rix[tid + s];
                }
            }
            __syncthreads();
        }
        if (tid == 0) {
            int j = rix[0];
            ts[t] = rv[0];
            tl[t] = (j % (NCLS - 1)) + 1;
            tb[t * 4 + 0] = cb[(long)j * 4 + 0];
            tb[t * 4 + 1] = cb[(long)j * 4 + 1];
            tb[t * 4 + 2] = cb[(long)j * 4 + 2];
            tb[t * 4 + 3] = cb[(long)j * 4 + 3];
            cs[j] = -2.0f;
        }
        __syncthreads();
    }

    // ---- class-offset boxes + areas ----
    for (int i = tid; i < KTOP; i += 256) {
        float off = (IMGSZ + 1.0f) * (float)tl[i];
        float x1 = tb[i * 4 + 0] + off, y1 = tb[i * 4 + 1] + off;
        float x2 = tb[i * 4 + 2] + off, y2 = tb[i * 4 + 3] + off;
        nbx[i * 4 + 0] = x1; nbx[i * 4 + 1] = y1;
        nbx[i * 4 + 2] = x2; nbx[i * 4 + 3] = y2;
        area[i] = (x2 - x1) * (y2 - y1);
        keep[i] = 0;
    }
    __syncthreads();

    // ---- greedy suppression ----
    for (int i = 0; i < KTOP; ++i) {
        if (tid == 0) sup = 0;
        __syncthreads();
        for (int j = tid; j < i; j += 256) {
            if (keep[j]) {
                float lx = fmaxf(nbx[i * 4 + 0], nbx[j * 4 + 0]);
                float ly = fmaxf(nbx[i * 4 + 1], nbx[j * 4 + 1]);
                float rx = fminf(nbx[i * 4 + 2], nbx[j * 4 + 2]);
                float ry = fminf(nbx[i * 4 + 3], nbx[j * 4 + 3]);
                float iw = fmaxf(rx - lx, 0.0f);
                float ih = fmaxf(ry - ly, 0.0f);
                float inter = iw * ih;
                float iou = inter / (area[i] + area[j] - inter);
                if (iou > NMS_TH) sup = 1;
            }
        }
        __syncthreads();
        if (tid == 0) keep[i] = (ts[i] > 0.0f) && !sup;
        __syncthreads();
    }

    // ---- compact first 100 into output ----
    if (tid == 0) {
        float* ob = out + (long)b * DET_PER * 4;
        float* os = out + (long)NB * DET_PER * 4 + (long)b * DET_PER;
        float* ol = out + (long)NB * DET_PER * 4 + (long)NB * DET_PER + (long)b * DET_PER;
        int r = 0;
        for (int i = 0; i < KTOP && r < DET_PER; ++i) {
            if (keep[i]) {
                ob[r * 4 + 0] = tb[i * 4 + 0];
                ob[r * 4 + 1] = tb[i * 4 + 1];
                ob[r * 4 + 2] = tb[i * 4 + 2];
                ob[r * 4 + 3] = tb[i * 4 + 3];
                os[r] = ts[i];
                ol[r] = (float)tl[i];
                ++r;
            }
        }
        for (; r < DET_PER; ++r) {
            ob[r * 4 + 0] = 0.0f; ob[r * 4 + 1] = 0.0f;
            ob[r * 4 + 2] = 0.0f; ob[r * 4 + 3] = 0.0f;
            os[r] = 0.0f; ol[r] = 0.0f;
        }
    }
}

// ---------------------------------------------------------------------------
// Launch
// ---------------------------------------------------------------------------
extern "C" void kernel_launch(void* const* d_in, const int* in_sizes, int n_in,
                              void* d_out, int out_size, void* d_ws, size_t ws_size,
                              hipStream_t stream) {
    const float* feats = (const float*)d_in[0];
    const float* props = (const float*)d_in[1];
    const float* w1    = (const float*)d_in[2];
    const float* b1    = (const float*)d_in[3];
    const float* w2    = (const float*)d_in[4];
    const float* b2    = (const float*)d_in[5];
    const float* wbox  = (const float*)d_in[6];
    const float* bbox  = (const float*)d_in[7];
    const float* wcls  = (const float*)d_in[8];
    const float* bcls  = (const float*)d_in[9];
    float* out = (float*)d_out;

    char* w = (char*)d_ws;
    auto alloc = [&](size_t bytes) -> char* {
        char* p = w;
        w += (bytes + 255) & ~(size_t)255;
        return p;
    };
    __bf16* Xbf   = (__bf16*)alloc((size_t)NROI * DIN * 2);
    __bf16* W1b   = (__bf16*)alloc((size_t)DIN * DH * 2);
    __bf16* W2b   = (__bf16*)alloc((size_t)DH * DH * 2);
    __bf16* Wboxb = (__bf16*)alloc((size_t)DH * NBOX_PAD * 2);
    __bf16* Wclsb = (__bf16*)alloc((size_t)DH * NCLS_PAD * 2);
    __bf16* H1    = (__bf16*)alloc((size_t)NROI * DH * 2);
    __bf16* H2    = (__bf16*)alloc((size_t)NROI * DH * 2);
    float*  BoxO  = (float*)alloc((size_t)NROI * NBOX_PAD * 4);
    float*  ClsO  = (float*)alloc((size_t)NROI * NCLS_PAD * 4);
    float*  bboxp = (float*)alloc((size_t)NBOX_PAD * 4);
    float*  bclsp = (float*)alloc((size_t)NCLS_PAD * 4);
    float*  candb = (float*)alloc((size_t)NB * NCAND * 4 * 4);
    float*  cands = (float*)alloc((size_t)NB * NCAND * 4);

    // 1. weight / bias conversion (bf16, N-padded)
    {
        long t1 = (long)DIN * DH;
        convert_pad_bf16<<<(t1 + 255) / 256, 256, 0, stream>>>(w1, W1b, DIN, DH, DH);
        long t2 = (long)DH * DH;
        convert_pad_bf16<<<(t2 + 255) / 256, 256, 0, stream>>>(w2, W2b, DH, DH, DH);
        long t3 = (long)DH * NBOX_PAD;
        convert_pad_bf16<<<(t3 + 255) / 256, 256, 0, stream>>>(wbox, Wboxb, DH, NBOX, NBOX_PAD);
        long t4 = (long)DH * NCLS_PAD;
        convert_pad_bf16<<<(t4 + 255) / 256, 256, 0, stream>>>(wcls, Wclsb, DH, NCLS, NCLS_PAD);
        pad_bias_f32<<<(NBOX_PAD + 255) / 256, 256, 0, stream>>>(bbox, bboxp, NBOX, NBOX_PAD);
        pad_bias_f32<<<(NCLS_PAD + 255) / 256, 256, 0, stream>>>(bcls, bclsp, NCLS, NCLS_PAD);
    }

    // 2. ROI max pool -> X (bf16)
    roi_pool<<<NROI * NBIN, 256, 0, stream>>>(feats, props, Xbf);

    // 3. FC1 (ReLU)  1024 x 12544 x 1024
    gemm_wmma_bf16<<<dim3(DH / 64, NROI / 64), 256, 0, stream>>>(
        Xbf, W1b, b1, nullptr, H1, NROI, DH, DIN, 1);

    // 4. FC2 (ReLU)  1024 x 1024 x 1024
    gemm_wmma_bf16<<<dim3(DH / 64, NROI / 64), 256, 0, stream>>>(
        H1, W2b, b2, nullptr, H2, NROI, DH, DH, 1);

    // 5. box / cls heads (f32 out)
    gemm_wmma_bf16<<<dim3(NBOX_PAD / 64, NROI / 64), 256, 0, stream>>>(
        H2, Wboxb, bboxp, BoxO, nullptr, NROI, NBOX_PAD, DH, 0);
    gemm_wmma_bf16<<<dim3(NCLS_PAD / 64, NROI / 64), 256, 0, stream>>>(
        H2, Wclsb, bclsp, ClsO, nullptr, NROI, NCLS_PAD, DH, 0);

    // 6. softmax + decode -> candidates
    decode_softmax<<<NROI, 128, 0, stream>>>(BoxO, ClsO, props, candb, cands);

    // 7. per-image top-k + NMS + compaction
    nms_image<<<NB, 256, 0, stream>>>(candb, cands, out);
}
